// StatefulLSTM_18150531793480
// MI455X (gfx1250) — matrix operand order repaired and checked by reading, not compiled, not run on hardware
//
#include <hip/hip_runtime.h>
#include <hip/hip_bf16.h>
#include <math.h>

// Problem constants (from reference): B=64, T=512, DI=512, H=1024
#define BB  64
#define TT  512
#define DI  512
#define HH  1024
#define G4  (4 * HH)      // 4096 gate columns
#define KC  (DI + HH)     // 1536 fused K dimension

typedef __attribute__((ext_vector_type(16))) __bf16 v16bf;
typedef __attribute__((ext_vector_type(8)))  __bf16 v8bf;
typedef __attribute__((ext_vector_type(8)))  float  v8f;

__device__ __forceinline__ float sigf(float x) { return 1.0f / (1.0f + __expf(-x)); }

__device__ __forceinline__ v16bf load_tile(const __bf16* p0, const __bf16* p1) {
  v16bf v;
  *((v8bf*)&v)     = *(const v8bf*)p0;
  *((v8bf*)&v + 1) = *(const v8bf*)p1;
  return v;
}

// ---------------------------------------------------------------------------
// One-time f32 -> bf16 conversion of the inputs (B,T,DI), row-major preserved.
// ---------------------------------------------------------------------------
__global__ void convert_x_kernel(const float* __restrict__ x, __bf16* __restrict__ xb, int n) {
  for (int i = blockIdx.x * blockDim.x + threadIdx.x; i < n; i += gridDim.x * blockDim.x)
    xb[i] = (__bf16)x[i];
}

// ---------------------------------------------------------------------------
// Build concatenated bf16 weight  Wc[g][0:512]=W_ih[g], Wc[g][512:1536]=W_hh[g]
// Row-major rows == WMMA B columns (K contiguous), so no transpose is needed.
// ---------------------------------------------------------------------------
__global__ void build_wcat_kernel(const float* __restrict__ Wih, const float* __restrict__ Whh,
                                  __bf16* __restrict__ Wc) {
  const int n = G4 * KC;
  for (int i = blockIdx.x * blockDim.x + threadIdx.x; i < n; i += gridDim.x * blockDim.x) {
    int g = i / KC, k = i - g * KC;
    float v = (k < DI) ? Wih[g * DI + k] : Whh[g * HH + (k - DI)];
    Wc[i] = (__bf16)v;
  }
}

__global__ void init_state_kernel(__bf16* __restrict__ h0b, float* __restrict__ c0) {
  int i = blockIdx.x * blockDim.x + threadIdx.x;
  if (i < BB * HH) { h0b[i] = (__bf16)0.0f; c0[i] = 0.0f; }
}

// ---------------------------------------------------------------------------
// One LSTM time step, fully fused:
//   gates(64 x 4096) = [x_t | h] (64 x 1536) @ Wc^T  + (b_ih + b_hh)
// Each wave owns one (16 batch x 16 H) tile and holds the 4 gate accumulators
// (i,f,g,o) in registers, then applies the cell update in-register and streams
// hs/cs (f32) out plus h_new (bf16, ping-pong) for the next step.
// Per K iteration all 10 b128 loads issue as one clause before the 4 WMMAs;
// the scheduler overlaps the next iteration's loads with the matrix pipe.
// 256 waves = 32 blocks x 256 threads.
// ---------------------------------------------------------------------------
__global__ __launch_bounds__(256)
void lstm_step_kernel(const __bf16* __restrict__ xb,    // (B,T,DI) bf16
                      const __bf16* __restrict__ Wc,    // (4H,KC)  bf16
                      const float*  __restrict__ b_ih,  // (4H)
                      const float*  __restrict__ b_hh,  // (4H)
                      const __bf16* __restrict__ h_in,  // (B,H) bf16  (step t-1)
                      __bf16*       __restrict__ h_out, // (B,H) bf16  (step t)
                      float*        __restrict__ c_st,  // (B,H) f32 in/out
                      float*        __restrict__ hs,    // (B,T,H) f32 out
                      float*        __restrict__ cs,    // (B,T,H) f32 out
                      int t)
{
  const int lane   = threadIdx.x & 31;
  const int wave   = (blockIdx.x * blockDim.x + threadIdx.x) >> 5;
  const int mt     = wave >> 6;      // 0..3  batch tile
  const int nt     = wave & 63;      // 0..63 hidden tile
  const int lane16 = lane & 15;
  const int hiA    = (lane >> 4) << 4;   // A: +16 K for upper half-wave
  const int hiB    = (lane >> 4) << 4;   // B: +16 K for upper half-wave

  // A-matrix rows (one per lane group): batch row = mt*16 + lane16
  const __bf16* xrow = xb   + ((size_t)(mt * 16 + lane16) * TT + t) * DI + hiA;
  const __bf16* hrow = h_in + (size_t)(mt * 16 + lane16) * HH + hiA;

  // B-matrix rows: 4 gate chunks; fold bias into the accumulators.
  const int g0 = nt * 16 + lane16;
  const __bf16* w0 = Wc + (size_t)(g0)          * KC + hiB;
  const __bf16* w1 = Wc + (size_t)(g0 + HH)     * KC + hiB;
  const __bf16* w2 = Wc + (size_t)(g0 + 2 * HH) * KC + hiB;
  const __bf16* w3 = Wc + (size_t)(g0 + 3 * HH) * KC + hiB;

  v8f acc0, acc1, acc2, acc3;
#pragma unroll
  for (int r = 0; r < 8; ++r) {
    acc0[r] = b_ih[g0]          + b_hh[g0];
    acc1[r] = b_ih[g0 + HH]     + b_hh[g0 + HH];
    acc2[r] = b_ih[g0 + 2 * HH] + b_hh[g0 + 2 * HH];
    acc3[r] = b_ih[g0 + 3 * HH] + b_hh[g0 + 3 * HH];
  }

  // ---- K loop over the fused [x_t | h] operand (DI % 32 == 0: no straddle) --
  // A 16x32 bf16 layout: lane<16 -> K 0..15; lane>=16 -> K 16..31 (contiguous)
  // B 32x16 bf16 layout: lane<16 -> K 0..15; lane>=16 -> K 16..31 (contiguous)
#pragma unroll 2
  for (int kk = 0; kk < KC; kk += 32) {
    const __bf16* arow = (kk < DI) ? (xrow + kk) : (hrow + (kk - DI));
    // Group all loads of this iteration ahead of the WMMAs (one clause).
    v16bf a  = load_tile(arow,          arow + 8);
    v16bf b0 = load_tile(w0 + kk,       w0 + kk + 8);
    v16bf b1 = load_tile(w1 + kk,       w1 + kk + 8);
    v16bf b2 = load_tile(w2 + kk,       w2 + kk + 8);
    v16bf b3 = load_tile(w3 + kk,       w3 + kk + 8);
    __builtin_prefetch(w0 + kk + 64, 0, 3);   // global_prefetch two blocks ahead
    __builtin_prefetch(w1 + kk + 64, 0, 3);
    __builtin_prefetch(w2 + kk + 64, 0, 3);
    __builtin_prefetch(w3 + kk + 64, 0, 3);
    acc0 = __builtin_amdgcn_wmma_f32_16x16x32_bf16(false, a, false, b0, (short)0, acc0, false, false);
    acc1 = __builtin_amdgcn_wmma_f32_16x16x32_bf16(false, a, false, b1, (short)0, acc1, false, false);
    acc2 = __builtin_amdgcn_wmma_f32_16x16x32_bf16(false, a, false, b2, (short)0, acc2, false, false);
    acc3 = __builtin_amdgcn_wmma_f32_16x16x32_bf16(false, a, false, b3, (short)0, acc3, false, false);
  }

  // Cell update for the 8 (row, col) elements this lane owns in the C layout:
  // VGPR r -> M = r (lanes 0-15) or 8+r (lanes 16-31); N = lane16.
  const int n     = nt * 16 + lane16;
  const int mbase = mt * 16 + ((lane < 16) ? 0 : 8);
#pragma unroll
  for (int r = 0; r < 8; ++r) {
    const int m = mbase + r;
    const float iv = sigf(acc0[r]);
    const float fv = sigf(acc1[r]);
    const float gv = tanhf(acc2[r]);
    const float ov = sigf(acc3[r]);
    const size_t sidx = (size_t)m * HH + n;
    const float cold = c_st[sidx];
    const float cnew = fv * cold + iv * gv;
    const float hnew = ov * tanhf(cnew);
    c_st[sidx] = cnew;
    const size_t oidx = ((size_t)m * TT + t) * HH + n;
    hs[oidx] = hnew;
    cs[oidx] = cnew;
    h_out[sidx] = (__bf16)hnew;
  }
}

// ---------------------------------------------------------------------------
// Gather h_last / c_last at t = length[b]-1.
// ---------------------------------------------------------------------------
__global__ void gather_last_kernel(const float* __restrict__ hs, const float* __restrict__ cs,
                                   const int* __restrict__ length,
                                   float* __restrict__ h_last, float* __restrict__ c_last) {
  int i = blockIdx.x * blockDim.x + threadIdx.x;
  if (i >= BB * HH) return;
  int b = i / HH, j = i - b * HH;
  int tl = length[b] - 1;
  size_t src = ((size_t)b * TT + tl) * HH + j;
  h_last[i] = hs[src];
  c_last[i] = cs[src];
}

extern "C" void kernel_launch(void* const* d_in, const int* in_sizes, int n_in,
                              void* d_out, int out_size, void* d_ws, size_t ws_size,
                              hipStream_t stream) {
  (void)in_sizes; (void)n_in; (void)out_size; (void)ws_size;

  const float* x   = (const float*)d_in[0];   // (B,T,DI)
  const float* Wih = (const float*)d_in[1];   // (4H,DI)
  const float* Whh = (const float*)d_in[2];   // (4H,H)
  const float* bih = (const float*)d_in[3];   // (4H)
  const float* bhh = (const float*)d_in[4];   // (4H)
  const int*   len = (const int*)d_in[5];     // (B)

  float* out    = (float*)d_out;
  float* hs     = out;                                  // (B,T,H)
  float* cs     = hs + (size_t)BB * TT * HH;            // (B,T,H)
  float* h_last = cs + (size_t)BB * TT * HH;            // (B,H)
  float* c_last = h_last + (size_t)BB * HH;             // (B,H)

  // Workspace carve-out (~45 MB total, 16B-aligned segments).
  char* ws = (char*)d_ws;
  size_t off = 0;
  __bf16* xb    = (__bf16*)(ws + off); off += (size_t)BB * TT * DI * 2;  // 32 MB
  __bf16* Wc    = (__bf16*)(ws + off); off += (size_t)G4 * KC * 2;       // 12 MB
  __bf16* hbuf0 = (__bf16*)(ws + off); off += (size_t)BB * HH * 2;       // 128 KB
  __bf16* hbuf1 = (__bf16*)(ws + off); off += (size_t)BB * HH * 2;       // 128 KB
  float*  c_st  = (float*)(ws + off);  off += (size_t)BB * HH * 4;       // 256 KB

  convert_x_kernel<<<2048, 256, 0, stream>>>(x, xb, BB * TT * DI);
  build_wcat_kernel<<<2048, 256, 0, stream>>>(Wih, Whh, Wc);
  init_state_kernel<<<(BB * HH + 255) / 256, 256, 0, stream>>>(hbuf0, c_st);

  for (int t = 0; t < TT; ++t) {
    const __bf16* hin = (t & 1) ? hbuf1 : hbuf0;
    __bf16*      hout = (t & 1) ? hbuf0 : hbuf1;
    lstm_step_kernel<<<32, 256, 0, stream>>>(xb, Wc, bih, bhh, hin, hout, c_st, hs, cs, t);
  }

  gather_last_kernel<<<(BB * HH + 255) / 256, 256, 0, stream>>>(hs, cs, len, h_last, c_last);
}
